// DeformConvNet_88519275971165
// MI455X (gfx1250) — matrix operand order repaired and checked by reading, not compiled, not run on hardware
//
#include <hip/hip_runtime.h>
#include <hip/hip_bf16.h>

typedef __attribute__((ext_vector_type(8)))  _Float16 v8h;
typedef __attribute__((ext_vector_type(16))) _Float16 v16h;
typedef __attribute__((ext_vector_type(8)))  float    v8f;

union V16 { v16h v; v8h h[2]; };

#define BATCH 4
#define HH 256
#define WW 256
#define HWPIX 65536
#define CIN 3
#define CMID 64
#define COFF 18

// ---------------------------------------------------------------------------
// Stage 0: pack offset_w / deform_w into WMMA A-operand layout (f16).
// A element order per lane used here (contiguous k, same as B): lanes<16 hold K {0..15},
// lanes>=16 hold K {16..31}; row M = mt*16 + lane%16.
// Global K ordering: K = tap*64 + ch  (ktile kt = tap*2 + half).
// apoff: [2 mtiles][18 kt][32 lanes][16 f16]; apdef: [4][18][32][16].
// ---------------------------------------------------------------------------
__global__ __launch_bounds__(256) void k_pack(const float* __restrict__ offw,
                                              const float* __restrict__ defw,
                                              _Float16* __restrict__ apoff,
                                              _Float16* __restrict__ apdef) {
  int idx = blockIdx.x * 256 + threadIdx.x;      // 0 .. 55295
  int which = (idx >= 18432);
  int li = which ? idx - 18432 : idx;
  int e    = li & 15;
  int lane = (li >> 4) & 31;
  int kt   = (li >> 9) % 18;
  int mt   = li / 9216;
  int ko = e + ((lane >= 16) ? 16 : 0);
  int Kg = kt * 32 + ko;                         // 0..575
  int tap = Kg >> 6;                             // K = tap*64 + ch
  int ch  = Kg & 63;
  int oc  = mt * 16 + (lane & 15);
  if (!which) {
    float v = (oc < COFF) ? offw[(oc * CMID + ch) * 9 + tap] : 0.f;
    apoff[li] = (_Float16)v;
  } else {
    float v = defw[(oc * CMID + ch) * 9 + tap];
    apdef[li] = (_Float16)v;
  }
}

__device__ __forceinline__ v16h load_a(const _Float16* __restrict__ ap,
                                       int mt, int kt, int lane) {
  const v8h* p = (const v8h*)(ap + (((size_t)(mt * 18 + kt) * 32 + lane) << 4));
  V16 a;
  a.h[0] = p[0];
  a.h[1] = p[1];
  return a.v;
}

// ---------------------------------------------------------------------------
// Stage 1: conv 3->64, pad 1, + bias.  Direct VALU conv (0.9 GFLOP, trivial).
// Writes feat as f16 channel-last [B][H][W][64] for WMMA B-operand loads.
// ---------------------------------------------------------------------------
__global__ __launch_bounds__(256) void k_conv1(const float* __restrict__ x,
                                               const float* __restrict__ w,
                                               const float* __restrict__ bias,
                                               _Float16* __restrict__ feat) {
  __shared__ float sw[CMID * 27];
  __shared__ float sb[CMID];
  int t = threadIdx.x;
  for (int i = t; i < CMID * 27; i += 256) sw[i] = w[i];
  if (t < CMID) sb[t] = bias[t];
  __syncthreads();

  int pix = blockIdx.x * 256 + t;                // 0 .. 262143 (exact grid)
  int b = pix >> 16;
  int p = pix & (HWPIX - 1);
  int y = p >> 8, xx = p & 255;

  float acc[CMID];
#pragma unroll
  for (int oc = 0; oc < CMID; ++oc) acc[oc] = sb[oc];

  for (int c = 0; c < CIN; ++c) {
#pragma unroll
    for (int di = 0; di < 3; ++di) {
#pragma unroll
      for (int dj = 0; dj < 3; ++dj) {
        int yy = y + di - 1, xc = xx + dj - 1;
        bool valid = (yy >= 0) & (yy < HH) & (xc >= 0) & (xc < WW);
        int yyc = min(max(yy, 0), HH - 1);
        int xcc = min(max(xc, 0), WW - 1);
        float v = x[((size_t)(b * CIN + c) << 16) + yyc * WW + xcc];
        v = valid ? v : 0.f;                      // unconditional load + select
        const float* wr = &sw[c * 9 + di * 3 + dj];
#pragma unroll
        for (int oc = 0; oc < CMID; ++oc) acc[oc] += wr[oc * 27] * v;
      }
    }
  }
  _Float16* o = feat + (size_t)pix * CMID;
#pragma unroll
  for (int oc = 0; oc < CMID; ++oc) o[oc] = (_Float16)acc[oc];
}

// ---------------------------------------------------------------------------
// Stage 2: offset conv 64->18 via WMMA. One wave = 16 pixels (row segment).
// M = out-channels (2 tiles, 18 padded to 32), N = 16 pixels, K = 9 taps * 64.
// Border taps: clamped (always-safe) address, UNCONDITIONAL load, then value
// select -> v_cndmask, no exec divergence, loads stay clause-grouped.
// Offsets stored f32 channel-last [B*H*W][18].
// ---------------------------------------------------------------------------
__global__ __launch_bounds__(256) void k_offconv(const _Float16* __restrict__ feat,
                                                 const _Float16* __restrict__ ap,
                                                 const float* __restrict__ bias,
                                                 float* __restrict__ off) {
  int gw = (blockIdx.x * 256 + threadIdx.x) >> 5;   // wave id, 0..16383 exact
  int lane = threadIdx.x & 31;
  int xg = gw & 15, y = (gw >> 4) & 255, b = gw >> 12;
  int px = lane & 15, hi = lane >> 4;
  int xx = (xg << 4) + px;

  v8f acc0 = {}, acc1 = {};
  const v8h zh = {};
  for (int tap = 0; tap < 9; ++tap) {
    int di = tap / 3 - 1, dj = tap % 3 - 1;
    int sy = y + di, sx = xx + dj;
    bool valid = (sy >= 0) & (sy < HH) & (sx >= 0) & (sx < WW);
    int syc = min(max(sy, 0), HH - 1);
    int sxc = min(max(sx, 0), WW - 1);
    const _Float16* fp = feat + ((size_t)((b << 16) + syc * WW + sxc) << 6);
#pragma unroll
    for (int half = 0; half < 2; ++half) {
      int kt = tap * 2 + half;
      int cb = half * 32 + hi * 16;               // B layout: contiguous 16 ch
      const v8h* q = (const v8h*)(fp + cb);
      v8h t0 = q[0];                              // hoisted: always executed
      v8h t1 = q[1];
      V16 bm;
      bm.h[0] = valid ? t0 : zh;                  // value select (cndmask)
      bm.h[1] = valid ? t1 : zh;
      v16h a0 = load_a(ap, 0, kt, lane);
      v16h a1 = load_a(ap, 1, kt, lane);
      acc0 = __builtin_amdgcn_wmma_f32_16x16x32_f16(false, a0, false, bm.v,
                                                    (short)0, acc0, false, false);
      acc1 = __builtin_amdgcn_wmma_f32_16x16x32_f16(false, a1, false, bm.v,
                                                    (short)0, acc1, false, false);
    }
  }
  float* op = off + (size_t)((b << 16) + y * WW + xx) * COFF;
  int cb0 = hi * 8;
#pragma unroll
  for (int r = 0; r < 8; ++r) {
    int ch = cb0 + r;                              // tile0: ch 0..15
    op[ch] = acc0[r] + bias[ch];
  }
  if (hi == 0) {                                   // tile1: only ch 16,17 real
    op[16] = acc1[0] + bias[16];
    op[17] = acc1[1] + bias[17];
  }
}

// ---------------------------------------------------------------------------
// Stage 3: deformable conv 64->64. Per tap: bilinear sample with corner
// weights computed in f32 (clamped addresses + validity-masked weights =
// reference semantics), blended with packed f16 FMAs, then 4 WMMA m-tiles
// against packed deform weights.
// ---------------------------------------------------------------------------
__global__ __launch_bounds__(256) void k_deform(const _Float16* __restrict__ feat,
                                                const float* __restrict__ off,
                                                const _Float16* __restrict__ ap,
                                                const float* __restrict__ bias,
                                                float* __restrict__ out) {
  int gw = (blockIdx.x * 256 + threadIdx.x) >> 5;   // 0..16383 exact
  int lane = threadIdx.x & 31;
  int xg = gw & 15, y = (gw >> 4) & 255, b = gw >> 12;
  int px = lane & 15, hi = lane >> 4;
  int xx = (xg << 4) + px;

  const float* offp = off + (size_t)((b << 16) + y * WW + xx) * COFF;
  v8f acc[4] = {{}, {}, {}, {}};

  for (int tap = 0; tap < 9; ++tap) {
    int di = tap / 3, dj = tap % 3;
    float oy = offp[2 * tap];
    float ox = offp[2 * tap + 1];
    float syf = (float)(y + di - 1) + oy;
    float sxf = (float)(xx + dj - 1) + ox;
    float y0f = floorf(syf), x0f = floorf(sxf);
    float fy = syf - y0f, fx = sxf - x0f;
    int y0 = (int)y0f, x0 = (int)x0f;

    _Float16 cw[4];
    size_t ca[4];
#pragma unroll
    for (int c = 0; c < 4; ++c) {
      int yc = y0 + (c >> 1);
      int xc = x0 + (c & 1);
      float wgt = ((c >> 1) ? fy : 1.f - fy) * ((c & 1) ? fx : 1.f - fx);
      bool valid = (yc >= 0) & (yc < HH) & (xc >= 0) & (xc < WW);
      cw[c] = (_Float16)(valid ? wgt : 0.f);
      int ycl = min(max(yc, 0), HH - 1);
      int xcl = min(max(xc, 0), WW - 1);
      ca[c] = (size_t)((b << 16) + ycl * WW + xcl) << 6;
    }
#pragma unroll
    for (int half = 0; half < 2; ++half) {
      int kt = tap * 2 + half;
      int cb = half * 32 + hi * 16;
      v16h bmv = {};
#pragma unroll
      for (int c = 0; c < 4; ++c) {
        const v8h* q = (const v8h*)(feat + ca[c] + cb);
        V16 g;
        g.h[0] = q[0];
        g.h[1] = q[1];
        v16h wv;
#pragma unroll
        for (int e = 0; e < 16; ++e) wv[e] = cw[c];
        bmv += g.v * wv;                           // v_pk_fma_f16 blend
      }
#pragma unroll
      for (int mt = 0; mt < 4; ++mt) {
        v16h a = load_a(ap, mt, kt, lane);
        acc[mt] = __builtin_amdgcn_wmma_f32_16x16x32_f16(false, a, false, bmv,
                                                         (short)0, acc[mt],
                                                         false, false);
      }
    }
  }
#pragma unroll
  for (int mt = 0; mt < 4; ++mt) {
#pragma unroll
    for (int r = 0; r < 8; ++r) {
      int ch = mt * 16 + r + 8 * hi;               // D layout: M = r + 8*(lane/16)
      out[((size_t)(b * 64 + ch) << 16) + y * WW + xx] = acc[mt][r] + bias[ch];
    }
  }
}

// ---------------------------------------------------------------------------
extern "C" void kernel_launch(void* const* d_in, const int* in_sizes, int n_in,
                              void* d_out, int out_size, void* d_ws, size_t ws_size,
                              hipStream_t stream) {
  (void)in_sizes; (void)n_in; (void)out_size; (void)ws_size;
  const float* x   = (const float*)d_in[0];
  const float* cw  = (const float*)d_in[1];
  const float* cb  = (const float*)d_in[2];
  const float* ow  = (const float*)d_in[3];
  const float* ob  = (const float*)d_in[4];
  const float* dw  = (const float*)d_in[5];
  const float* db  = (const float*)d_in[6];
  float* out = (float*)d_out;

  char* ws = (char*)d_ws;
  _Float16* feat  = (_Float16*)ws;                                  // 33554432 B
  float*    off   = (float*)(ws + 33554432);                        // 18874368 B
  _Float16* apoff = (_Float16*)(ws + 33554432 + 18874368);          //    36864 B
  _Float16* apdef = (_Float16*)(ws + 33554432 + 18874368 + 36864);  //    73728 B

  k_pack   <<<216,  256, 0, stream>>>(ow, dw, apoff, apdef);
  k_conv1  <<<1024, 256, 0, stream>>>(x, cw, cb, feat);
  k_offconv<<<2048, 256, 0, stream>>>(feat, apoff, ob, off);
  k_deform <<<2048, 256, 0, stream>>>(feat, off, apdef, db, out);
}
